// AutoCorrelation_40089224741150
// MI455X (gfx1250) — matrix operand order repaired and checked by reading, not compiled, not run on hardware
//
#include <hip/hip_runtime.h>

// Problem constants (from reference setup_inputs)
#define B_   8
#define L_   3072
#define H_   8
#define E_   64
#define C_   512            // H*E, contiguous innermost in memory
#define NT_  192            // L/16 row tiles
#define TOPK 40             // int(5*ln(3072))

typedef __attribute__((ext_vector_type(16))) __bf16 v16bf;
typedef __attribute__((ext_vector_type(8)))  __bf16 v8bf;
typedef __attribute__((ext_vector_type(8)))  float  v8f;

#if defined(__AMDGCN__) && __has_builtin(__builtin_amdgcn_global_load_async_to_lds_b128)
#define USE_ASYNC_LDS 1
typedef int v4i_ __attribute__((vector_size(16)));
typedef v4i_ __attribute__((address_space(1))) as1_v4i_t;
typedef v4i_ __attribute__((address_space(3))) as3_v4i_t;
#endif

__device__ __forceinline__ unsigned short f2bf(float f) {
    unsigned int u = __float_as_uint(f);
    unsigned int r = u + 0x7FFFu + ((u >> 16) & 1u);   // round-to-nearest-even
    return (unsigned short)(r >> 16);
}

// ---------------- Kernel 1: f32 -> bf16 conversion of Q and K ----------------
__global__ __launch_bounds__(256)
void cvt_bf16_kernel(const float* __restrict__ q, const float* __restrict__ k,
                     unsigned short* __restrict__ qb, unsigned short* __restrict__ kb,
                     int n4) {
    int i = blockIdx.x * 256 + threadIdx.x;
    if (i >= n4) return;
    const float4* src = (const float4*)(blockIdx.y ? k : q);
    uint2* dst = (uint2*)(blockIdx.y ? kb : qb);
    float4 v = src[i];
    unsigned int lo = (unsigned int)f2bf(v.x) | ((unsigned int)f2bf(v.y) << 16);
    unsigned int hi = (unsigned int)f2bf(v.z) | ((unsigned int)f2bf(v.w) << 16);
    dst[i] = make_uint2(lo, hi);
}

// ---------------- Kernel 2: zero the correlation accumulator ----------------
__global__ __launch_bounds__(256)
void zero_kernel(float* __restrict__ p, int n) {
    int i = blockIdx.x * 256 + threadIdx.x;
    if (i < n) p[i] = 0.0f;
}

// ---------------- Kernel 3: banded Gram correlation via WMMA ----------------
// mean[b,t] += wrapped-diagonal sums of Q_bf16 * K_bf16^T
// grid.x = B_ * (NT_/4) * S_CHUNKS ; 128 threads = 4 waves, one band P per wave.
#define S_CHUNKS     8
#define S_PER_CHUNK  (NT_ / S_CHUNKS)   // 24
#define P_PER_WG     4

__global__ __launch_bounds__(128)
void corr_wmma_kernel(const unsigned short* __restrict__ qb,
                      const unsigned short* __restrict__ kb,
                      float* __restrict__ mean) {
    const int wg     = blockIdx.x;
    const int schunk = wg % S_CHUNKS;
    const int pgrp   = (wg / S_CHUNKS) % (NT_ / P_PER_WG);
    const int b      = wg / (S_CHUNKS * (NT_ / P_PER_WG));
    const int wave   = threadIdx.x >> 5;
    const int lane   = threadIdx.x & 31;
    const int P      = pgrp * P_PER_WG + wave;   // diagonal band index [0,192)
    const int lr     = lane & 15;                // row-in-tile / col-in-tile
    const int lh     = lane >> 4;                // K-half selector

    __shared__ __align__(64) unsigned short kt[16 * C_];  // 16 KB K-tile stage

    const unsigned short* qbase = qb + (size_t)b * L_ * C_;
    const unsigned short* kbase = kb + (size_t)b * L_ * C_;

    v8f acc = {0.f, 0.f, 0.f, 0.f, 0.f, 0.f, 0.f, 0.f};

    for (int si = 0; si < S_PER_CHUNK; ++si) {
        const int S = schunk * S_PER_CHUNK + si;

        __syncthreads();  // protect kt against readers from previous iteration
#if defined(USE_ASYNC_LDS)
        {   // async DMA-style stage: global -> LDS, no VGPR round-trip (ASYNCcnt)
            as1_v4i_t* gsrc = (as1_v4i_t*)(kbase + (size_t)(16 * S) * C_);
            as3_v4i_t* ldst = (as3_v4i_t*)kt;
            const int off16 = (int)threadIdx.x;      // in 16-byte units
            #pragma unroll
            for (int i = 0; i < 8; ++i)
                __builtin_amdgcn_global_load_async_to_lds_b128(
                    gsrc + off16 + i * 128, ldst + off16 + i * 128, 0, 0);
#if __has_builtin(__builtin_amdgcn_s_wait_asynccnt)
            __builtin_amdgcn_s_wait_asynccnt(0);
#else
            asm volatile("s_wait_asynccnt 0x0" ::: "memory");
#endif
        }
#else
        {   // cooperative global->LDS stage of K rows [16S, 16S+16) x 512 ch
            const uint4* src = (const uint4*)(kbase + (size_t)(16 * S) * C_);
            uint4* dst = (uint4*)kt;
            #pragma unroll
            for (int i = 0; i < (16 * C_ * 2 / 16) / 128; ++i)    // 8 iters
                dst[threadIdx.x + i * 128] = src[threadIdx.x + i * 128];
        }
#endif
        __syncthreads();

        if (si + 1 < S_PER_CHUNK)   // hint next K tile toward caches
            __builtin_prefetch(kbase + (size_t)(16 * (S + 1)) * C_, 0, 1);

        int ap = S + P; if (ap >= NT_) ap -= NT_;
        const unsigned short* arow = qbase + (size_t)(ap * 16 + lr) * C_;

        #pragma unroll 8
        for (int c = 0; c < C_ / 32; ++c) {
            const int cb = c * 32;
            // A operand (16x32 bf16): lane lr = row M, lh selects K halves
            v8bf alo = *(const v8bf*)(arow + cb + lh * 8);        // K 0..7 / 8..15
            v8bf ahi = *(const v8bf*)(arow + cb + 16 + lh * 8);   // K 16..23 / 24..31
            v16bf A = __builtin_shufflevector(alo, ahi,
                0, 1, 2, 3, 4, 5, 6, 7, 8, 9, 10, 11, 12, 13, 14, 15);
            // B operand (32x16 bf16): lane lr = col N, lh selects K halves (same order as A)
            v8bf blo = *(const v8bf*)(kt + lr * C_ + cb + lh * 8);
            v8bf bhi = *(const v8bf*)(kt + lr * C_ + cb + 16 + lh * 8);
            v16bf Bv = __builtin_shufflevector(blo, bhi,
                0, 1, 2, 3, 4, 5, 6, 7, 8, 9, 10, 11, 12, 13, 14, 15);

            acc = __builtin_amdgcn_wmma_f32_16x16x32_bf16(
                false, A, false, Bv, (short)0, acc, false, false);
        }
    }

    // Diagonal scatter: D[m,n] contributes to t = (16P + m - n) mod L
    const int tb = 16 * P - lr;
    float* mrow = mean + (size_t)b * L_;
    #pragma unroll
    for (int r = 0; r < 8; ++r) {
        const int m = r + 8 * lh;
        int t = tb + m;
        if (t < 0)    t += L_;
        if (t >= L_)  t -= L_;
        atomicAdd(mrow + t, acc[r]);
    }
}

// ---------------- Kernel 4: per-batch top-40 + softmax ----------------
__global__ __launch_bounds__(256)
void topk_kernel(const float* __restrict__ mean,
                 float* __restrict__ wts, int* __restrict__ shifts) {
    const int b = blockIdx.x;
    const int tid = threadIdx.x;
    __shared__ float vals[L_];
    __shared__ float rv[256];
    __shared__ int   ri[256];
    __shared__ float topv[TOPK];
    __shared__ int   topi[TOPK];

    const float scale = 1.0f / (float)C_;   // mean over H and E
    for (int i = tid; i < L_; i += 256) vals[i] = mean[(size_t)b * L_ + i] * scale;
    __syncthreads();

    for (int k = 0; k < TOPK; ++k) {
        float bv = -3.0e38f; int bi = 0;
        for (int i = tid; i < L_; i += 256)
            if (vals[i] > bv) { bv = vals[i]; bi = i; }
        rv[tid] = bv; ri[tid] = bi;
        __syncthreads();
        for (int off = 128; off > 0; off >>= 1) {
            if (tid < off && rv[tid + off] > rv[tid]) {
                rv[tid] = rv[tid + off]; ri[tid] = ri[tid + off];
            }
            __syncthreads();
        }
        if (tid == 0) { topv[k] = rv[0]; topi[k] = ri[0]; vals[ri[0]] = -3.0e38f; }
        __syncthreads();
    }

    if (tid == 0) {
        float mx = topv[0];                // descending -> first is max
        float e[TOPK], sum = 0.0f;
        for (int k = 0; k < TOPK; ++k) { e[k] = __expf(topv[k] - mx); sum += e[k]; }
        float inv = 1.0f / sum;
        for (int k = 0; k < TOPK; ++k) wts[b * TOPK + k] = e[k] * inv;
        if (b == 0) for (int k = 0; k < TOPK; ++k) shifts[k] = topi[k];
    }
}

// ---------------- Kernel 5: weighted circular-shift aggregation of V ----------------
__global__ __launch_bounds__(256)
void agg_kernel(const float* __restrict__ V, const float* __restrict__ wts,
                const int* __restrict__ shifts, float* __restrict__ out) {
    int idx = blockIdx.x * 256 + threadIdx.x;          // over B*L*(C/4) float4s
    const int n4 = B_ * L_ * (C_ / 4);
    if (idx >= n4) return;
    const int c4 = idx & (C_ / 4 - 1);
    const int bl = idx >> 7;                           // C_/4 == 128
    const int b  = bl / L_;
    const int l  = bl - b * L_;

    const float4* v4 = (const float4*)V + (size_t)b * L_ * (C_ / 4);
    float ax = 0.f, ay = 0.f, az = 0.f, aw = 0.f;
    #pragma unroll 8
    for (int k = 0; k < TOPK; ++k) {
        int s = l + shifts[k]; if (s >= L_) s -= L_;
        const float w = wts[b * TOPK + k];
        float4 x = v4[(size_t)s * (C_ / 4) + c4];
        ax += w * x.x; ay += w * x.y; az += w * x.z; aw += w * x.w;
    }
    float4 o; o.x = ax; o.y = ay; o.z = az; o.w = aw;
    ((float4*)out)[idx] = o;
}

// ---------------- Host entry ----------------
extern "C" void kernel_launch(void* const* d_in, const int* in_sizes, int n_in,
                              void* d_out, int out_size, void* d_ws, size_t ws_size,
                              hipStream_t stream) {
    const float* q = (const float*)d_in[0];
    const float* k = (const float*)d_in[1];
    const float* v = (const float*)d_in[2];
    (void)in_sizes; (void)n_in; (void)ws_size; (void)out_size;

    const size_t nQK = (size_t)B_ * L_ * C_;           // 12.58M elements
    unsigned short* qb = (unsigned short*)d_ws;
    unsigned short* kb = qb + nQK;
    float* mean   = (float*)(kb + nQK);
    float* wts    = mean + B_ * L_;
    int*   shifts = (int*)(wts + B_ * TOPK);

    // 1) convert Q,K to bf16
    {
        const int n4 = (int)(nQK / 4);                 // 3,145,728
        dim3 grid((n4 + 255) / 256, 2);
        cvt_bf16_kernel<<<grid, 256, 0, stream>>>(q, k, qb, kb, n4);
    }
    // 2) zero mean accumulator
    {
        const int n = B_ * L_;
        zero_kernel<<<(n + 255) / 256, 256, 0, stream>>>(mean, n);
    }
    // 3) WMMA banded correlation
    {
        const int grid = B_ * (NT_ / P_PER_WG) * S_CHUNKS;   // 3072 WGs
        corr_wmma_kernel<<<grid, 128, 0, stream>>>(qb, kb, mean);
    }
    // 4) top-40 + softmax per batch
    topk_kernel<<<B_, 256, 0, stream>>>(mean, wts, shifts);
    // 5) weighted roll-gather of V
    {
        const int n4 = B_ * L_ * (C_ / 4);
        agg_kernel<<<(n4 + 255) / 256, 256, 0, stream>>>(v, wts, shifts, (float*)d_out);
    }
}
